// Parton_34806414966867
// MI455X (gfx1250) — matrix operand order repaired and checked by reading, not compiled, not run on hardware
//
#include <hip/hip_runtime.h>
#include <hip/hip_bf16.h>

#define NEL     256
#define NPAIR   32640          // NEL*(NEL-1)/2
#define MROWS   65280          // 2*NPAIR
#define MTILES  4080           // MROWS/16
#define DDIM    256
#define NHEAD   4
#define DHEAD   64
#define ML2INV  100.0f         // 1/(0.1^2)

typedef float  v8f   __attribute__((ext_vector_type(8)));
typedef __bf16 bf16_t;
typedef bf16_t v16bf __attribute__((ext_vector_type(16)));

union FragB { uint4 q[2]; v16bf v; };
union FragC { v8f v; float f[8]; };

// ---------- helpers ----------
__device__ inline unsigned short f2bf(float x) {
    unsigned u = __float_as_uint(x);
    unsigned r = (u + 0x7FFFu + ((u >> 16) & 1u)) >> 16;
    return (unsigned short)r;
}

__device__ inline int paircum(int i) { return (i * (2 * NEL - 1 - i)) >> 1; }

__device__ inline void pairij(int p, int& i, int& j) {
    float a = (float)(2 * NEL - 1);
    int ig = (int)floorf((a - sqrtf(fmaxf(a * a - 8.0f * (float)p, 0.f))) * 0.5f);
    if (ig < 0) ig = 0;
    if (ig > NEL - 2) ig = NEL - 2;
    while (ig < NEL - 2 && paircum(ig + 1) <= p) ++ig;
    while (ig > 0 && paircum(ig) > p) --ig;
    i = ig;
    j = ig + 1 + (p - paircum(ig));
}

// ---------- weight prepack: fp32 [256][256] (k,n) -> bf16 WMMA B-fragment layout ----------
// Bpacked index: ((nt*8+kt)*32 + lane)*16 + jj ; lane0-15: N=lane, K=kt*32+jj ;
//                lanes16-31: N=lane-16, K=kt*32+16+jj
__global__ __launch_bounds__(256) void Parton_pack_kernel(const float* __restrict__ W,
                                                          unsigned short* __restrict__ out) {
    int b = blockIdx.x;            // 0..127 = nt*8+kt
    int nt = b >> 3, kt = b & 7;
    int tid = threadIdx.x;
    for (int e = 0; e < 2; ++e) {
        int idx = tid + 256 * e;   // 0..511
        int lane = idx >> 4, jj = idx & 15;
        int n = nt * 16 + (lane & 15);
        int kk = kt * 32 + ((lane < 16) ? 0 : 16) + jj;
        out[((size_t)b * 32 + lane) * 16 + jj] = f2bf(W[(size_t)kk * DDIM + n]);
    }
}

// ---------- input features: h0 = feat @ W_in ----------
__global__ __launch_bounds__(256) void Parton_feat_kernel(const float* __restrict__ elec,
                                                          const float* __restrict__ W_in,
                                                          float* __restrict__ h) {
    int m = blockIdx.x, c = threadIdx.x;
    int p = m >> 1, s = m & 1;
    int i, j; pairij(p, i, j);
    int e = s ? j : i;
    float th = elec[e * 2], ph = elec[e * 2 + 1];
    float st = sinf(th);
    float f0 = cosf(th), f1 = st * cosf(ph), f2 = st * sinf(ph);
    h[(size_t)m * DDIM + c] = f0 * W_in[c] + f1 * W_in[DDIM + c] + f2 * W_in[2 * DDIM + c];
}

// ---------- fused WMMA GEMM: out = epilogue(A[M,256] @ B[256,256]) ----------
// mode 0: out = acc + bias
// mode 1: out = LN(res + acc)                    (scale=lnS, shift=lnB)
// mode 2: out = LN(res + tanh(acc + bias))
__global__ __launch_bounds__(256) void Parton_gemm_kernel(const float* __restrict__ A,
                                                          const unsigned short* __restrict__ Bp,
                                                          const float* __restrict__ bias,
                                                          const float* __restrict__ res,
                                                          const float* __restrict__ lnS,
                                                          const float* __restrict__ lnB,
                                                          float* __restrict__ out,
                                                          int mode) {
    __shared__ unsigned short ldsA[16 * 264];  // bf16 A tile, padded stride
    __shared__ float ldsT[16 * 260];           // fp32 result tile for LN epilogue
    __shared__ float sMean[16], sRstd[16];

    const int tid  = threadIdx.x;
    const int lane = tid & 31;
    const int wave = tid >> 5;
    const size_t m0 = (size_t)blockIdx.x * 16;

    // stage A tile fp32 -> bf16 into LDS (thread = column, loop = row)
    for (int i = 0; i < 16; ++i)
        ldsA[i * 264 + tid] = f2bf(A[(m0 + i) * DDIM + tid]);
    __syncthreads();

    const int nt0 = wave * 2, nt1 = nt0 + 1;
    const int arow = lane & 15;
    const unsigned short* aRow = &ldsA[arow * 264];

    v8f c0 = {};
    v8f c1 = {};

#pragma unroll
    for (int kt = 0; kt < 8; ++kt) {
        // A fragment: lanes take K=[kb,kb+16) with kb=kt*32 (+16 for upper half)
        int kb = kt * 32 + ((lane >> 4) << 4);
        FragB a;
        a.q[0] = *(const uint4*)(aRow + kb);
        a.q[1] = *(const uint4*)(aRow + kb + 8);

        const uint4* pb0 = (const uint4*)(Bp + (((size_t)(nt0 * 8 + kt)) * 32 + lane) * 16);
        const uint4* pb1 = (const uint4*)(Bp + (((size_t)(nt1 * 8 + kt)) * 32 + lane) * 16);
        __builtin_prefetch((const void*)(Bp + (((size_t)(nt0 * 8 + ((kt + 1) & 7))) * 32 + lane) * 16), 0, 0);
        FragB b0, b1;
        b0.q[0] = pb0[0]; b0.q[1] = pb0[1];
        b1.q[0] = pb1[0]; b1.q[1] = pb1[1];

        c0 = __builtin_amdgcn_wmma_f32_16x16x32_bf16(false, a.v, false, b0.v, (short)0, c0, false, false);
        c1 = __builtin_amdgcn_wmma_f32_16x16x32_bf16(false, a.v, false, b1.v, (short)0, c1, false, false);
    }

    FragC f0, f1; f0.v = c0; f1.v = c1;
    const int colL = lane & 15;
    const int rOff = (lane >> 4) * 8;   // C layout: VGPR r -> row r (lanes 0-15) / r+8 (lanes 16-31)

    if (mode == 0) {
#pragma unroll
        for (int r = 0; r < 8; ++r) {
            int row = r + rOff;
            int n0 = nt0 * 16 + colL, n1 = nt1 * 16 + colL;
            float b0v = bias ? bias[n0] : 0.f;
            float b1v = bias ? bias[n1] : 0.f;
            out[(m0 + row) * DDIM + n0] = f0.f[r] + b0v;
            out[(m0 + row) * DDIM + n1] = f1.f[r] + b1v;
        }
    } else {
#pragma unroll
        for (int r = 0; r < 8; ++r) {
            int row = r + rOff;
            ldsT[row * 260 + nt0 * 16 + colL] = f0.f[r];
            ldsT[row * 260 + nt1 * 16 + colL] = f1.f[r];
        }
        __syncthreads();
        // t = res + f(acc); thread = column, loop = row
        for (int i = 0; i < 16; ++i) {
            float a  = ldsT[i * 260 + tid];
            float rv = res[(m0 + i) * DDIM + tid];
            float t  = (mode == 1) ? (rv + a)
                                   : (rv + tanhf(a + (bias ? bias[tid] : 0.f)));
            ldsT[i * 260 + tid] = t;
        }
        __syncthreads();
        if (tid < 16) {
            float s = 0.f, ss = 0.f;
            for (int c = 0; c < DDIM; ++c) { float x = ldsT[tid * 260 + c]; s += x; ss += x * x; }
            float mean = s * (1.f / DDIM);
            float var  = ss * (1.f / DDIM) - mean * mean;
            sMean[tid] = mean;
            sRstd[tid] = rsqrtf(var + 1e-5f);
        }
        __syncthreads();
        float sc = lnS[tid], sh = lnB[tid];
        for (int i = 0; i < 16; ++i)
            out[(m0 + i) * DDIM + tid] = (ldsT[i * 260 + tid] - sMean[i]) * sRstd[i] * sc + sh;
    }
}

// ---------- 2-token attention (per pair, per head) ----------
__global__ __launch_bounds__(256) void Parton_attn_kernel(const float* __restrict__ q,
                                                          const float* __restrict__ k,
                                                          const float* __restrict__ v,
                                                          float* __restrict__ o) {
    int idx = blockIdx.x * 256 + threadIdx.x;
    if (idx >= NPAIR * NHEAD) return;
    int p = idx >> 2, h = idx & 3;
    size_t r0 = (size_t)(2 * p) * DDIM + h * DHEAD;
    size_t r1 = r0 + DDIM;
    float s00 = 0, s01 = 0, s10 = 0, s11 = 0;
    for (int d = 0; d < DHEAD; ++d) {
        float a0 = q[r0 + d], a1 = q[r1 + d], b0 = k[r0 + d], b1 = k[r1 + d];
        s00 += a0 * b0; s01 += a0 * b1; s10 += a1 * b0; s11 += a1 * b1;
    }
    s00 *= 0.125f; s01 *= 0.125f; s10 *= 0.125f; s11 *= 0.125f;
    float m0 = fmaxf(s00, s01), e00 = __expf(s00 - m0), e01 = __expf(s01 - m0);
    float z0 = e00 + e01, a00 = e00 / z0, a01 = e01 / z0;
    float m1 = fmaxf(s10, s11), e10 = __expf(s10 - m1), e11 = __expf(s11 - m1);
    float z1 = e10 + e11, a10 = e10 / z1, a11 = e11 / z1;
    for (int d = 0; d < DHEAD; ++d) {
        float v0 = v[r0 + d], v1 = v[r1 + d];
        o[r0 + d] = a00 * v0 + a01 * v1;
        o[r1 + d] = a10 * v0 + a11 * v1;
    }
}

// ---------- pair orbitals: 2x2 complex det * trunc^2 ----------
__global__ __launch_bounds__(256) void Parton_orb_kernel(const float* __restrict__ h,
                                                         const float* __restrict__ Wr,
                                                         const float* __restrict__ Wi,
                                                         const float* __restrict__ elec,
                                                         float2* __restrict__ porb) {
    int p = blockIdx.x * 256 + threadIdx.x;
    if (p >= NPAIR) return;
    const float* h0 = h + (size_t)(2 * p) * DDIM;
    const float* h1 = h0 + DDIM;
    float r00 = 0, r01 = 0, r10 = 0, r11 = 0, i00 = 0, i01 = 0, i10 = 0, i11 = 0;
    for (int c = 0; c < DDIM; ++c) {
        float x0 = h0[c], x1 = h1[c];
        float wr0 = Wr[2 * c], wr1 = Wr[2 * c + 1];
        float wi0 = Wi[2 * c], wi1 = Wi[2 * c + 1];
        r00 += x0 * wr0; r01 += x0 * wr1; r10 += x1 * wr0; r11 += x1 * wr1;
        i00 += x0 * wi0; i01 += x0 * wi1; i10 += x1 * wi0; i11 += x1 * wi1;
    }
    float dr = (r00 * r11 - i00 * i11) - (r01 * r10 - i01 * i10);
    float di = (r00 * i11 + i00 * r11) - (r01 * i10 + i01 * r10);
    int a, b; pairij(p, a, b);
    float tha = elec[a * 2], pha = elec[a * 2 + 1];
    float thb = elec[b * 2], phb = elec[b * 2 + 1];
    float ca = cosf(0.5f * tha), sa = sinf(0.5f * tha);
    float cb = cosf(0.5f * thb), sb = sinf(0.5f * thb);
    float d2 = 0.5f * (pha - phb);
    float re = (ca * sb - sa * cb) * cosf(d2);
    float im = (ca * sb + sa * cb) * sinf(d2);
    float chord2 = re * re + im * im;
    float tr = 1.f - __expf(-chord2 * ML2INV);
    float t2 = tr * tr;
    porb[p] = make_float2(t2 * dr, t2 * di);
}

// ---------- pf+cusp assembly + bosonic log partial sums ----------
__global__ __launch_bounds__(256) void Parton_assemble_kernel(const float* __restrict__ elec,
                                                              const float2* __restrict__ porb,
                                                              float2* __restrict__ A,
                                                              float2* __restrict__ partials) {
    __shared__ float sL[256], sA[256];
    int tid = threadIdx.x;
    int idx = blockIdx.x * 256 + tid;
    int i = idx >> 8, j = idx & 255;
    float thi = elec[i * 2], phi_ = elec[i * 2 + 1];
    float thj = elec[j * 2], phj  = elec[j * 2 + 1];
    float ci = cosf(0.5f * thi), si = sinf(0.5f * thi);
    float cj = cosf(0.5f * thj), sj = sinf(0.5f * thj);
    float d = 0.5f * (phi_ - phj);
    float er = (ci * sj - si * cj) * cosf(d);
    float ei = (ci * sj + si * cj) * sinf(d);
    float eye = (i == j) ? 1.f : 0.f;
    float ecr = er + eye, eci = ei;
    float rc2 = ecr * ecr + eci * eci;
    float g = __expf(-rc2 * ML2INV);
    float cr = ecr * g, cy = eci * g;
    float pfr = 0.f, pfi = 0.f;
    if (i < j)      { float2 t = porb[paircum(i) + (j - i - 1)]; pfr =  t.x; pfi =  t.y; }
    else if (i > j) { float2 t = porb[paircum(j) + (i - j - 1)]; pfr = -t.x; pfi = -t.y; }
    A[idx] = make_float2(pfr + cr, pfi + cy);
    float lm = 0.f, am = 0.f;
    if (i != j) {
        float rho2 = er * er + ei * ei;
        float rho = sqrtf(rho2);
        float f = sqrtf(0.01f + rho2) / rho;
        float mr = f * er, mi = f * ei;
        lm = 0.5f * logf(mr * mr + mi * mi);
        am = atan2f(mi, mr);
    }
    sL[tid] = lm; sA[tid] = am;
    __syncthreads();
    for (int s = 128; s > 0; s >>= 1) {
        if (tid < s) { sL[tid] += sL[tid + s]; sA[tid] += sA[tid + s]; }
        __syncthreads();
    }
    if (tid == 0) partials[blockIdx.x] = make_float2(sL[0], sA[0]);
}

// ---------- single-workgroup complex LU with partial pivoting -> 0.5*slogdet ----------
__global__ __launch_bounds__(256) void Parton_lu_kernel(float2* __restrict__ A, float* __restrict__ lf) {
    __shared__ float2 prow[NEL];
    __shared__ float red[256];
    __shared__ int redi[256];
    int t = threadIdx.x;
    float logabs = 0.f, ang = 0.f;
    int swaps = 0;
    for (int kk = 0; kk < NEL; ++kk) {
        float cand = -1.f;
        if (t >= kk) { float2 a = A[t * NEL + kk]; cand = a.x * a.x + a.y * a.y; }
        red[t] = cand; redi[t] = t;
        __syncthreads();
        for (int s = 128; s > 0; s >>= 1) {
            if (t < s && red[t + s] > red[t]) { red[t] = red[t + s]; redi[t] = redi[t + s]; }
            __syncthreads();
        }
        int piv = redi[0];
        __syncthreads();
        if (piv != kk) {
            float2 a = A[kk * NEL + t], b = A[piv * NEL + t];
            A[kk * NEL + t] = b; A[piv * NEL + t] = a;
        }
        __syncthreads();
        prow[t] = A[kk * NEL + t];
        __syncthreads();
        float2 pv = prow[kk];
        if (t == 0) {
            logabs += 0.5f * logf(pv.x * pv.x + pv.y * pv.y);
            ang += atan2f(pv.y, pv.x);
            if (piv != kk) swaps ^= 1;
        }
        if (t > kk) {
            float2 a = A[t * NEL + kk];
            float dd = pv.x * pv.x + pv.y * pv.y;
            float mr = (a.x * pv.x + a.y * pv.y) / dd;
            float mi = (a.y * pv.x - a.x * pv.y) / dd;
            A[t * NEL + kk] = make_float2(mr, mi);
            for (int c = kk + 1; c < NEL; ++c) {
                float2 u = prow[c];
                float2 w = A[t * NEL + c];
                w.x -= mr * u.x - mi * u.y;
                w.y -= mr * u.y + mi * u.x;
                A[t * NEL + c] = w;
            }
        }
        __syncthreads();
    }
    if (t == 0) {
        if (swaps) ang += 3.14159265358979323846f;
        float tp = 6.28318530717958647692f;
        ang = fmodf(ang, tp);
        if (ang >  3.14159265358979f) ang -= tp;
        if (ang <= -3.14159265358979f) ang += tp;
        lf[0] = 0.5f * logabs;
        lf[1] = 0.5f * ang;
    }
}

// ---------- final deterministic reduction ----------
__global__ __launch_bounds__(256) void Parton_final_kernel(const float2* __restrict__ partials,
                                                           const float* __restrict__ lf,
                                                           float* __restrict__ out, int out_size) {
    __shared__ float sL[256], sA[256];
    int t = threadIdx.x;
    float2 p = partials[t];
    sL[t] = p.x; sA[t] = p.y;
    __syncthreads();
    for (int s = 128; s > 0; s >>= 1) {
        if (t < s) { sL[t] += sL[t + s]; sA[t] += sA[t + s]; }
        __syncthreads();
    }
    if (t == 0) {
        out[0] = lf[0] + sL[0];
        if (out_size > 1) out[1] = lf[1] + sA[0];
    }
}

// ---------- host ----------
extern "C" void kernel_launch(void* const* d_in, const int* in_sizes, int n_in,
                              void* d_out, int out_size, void* d_ws, size_t ws_size,
                              hipStream_t stream) {
    (void)in_sizes; (void)n_in; (void)ws_size;
    const float* elec   = (const float*)d_in[0];
    const float* W_in   = (const float*)d_in[1];
    const float* Wq     = (const float*)d_in[2];
    const float* bq     = (const float*)d_in[3];
    const float* Wk     = (const float*)d_in[4];
    const float* bk     = (const float*)d_in[5];
    const float* Wv     = (const float*)d_in[6];
    const float* bv     = (const float*)d_in[7];
    const float* Wo     = (const float*)d_in[8];
    const float* bo     = (const float*)d_in[9];
    const float* W1     = (const float*)d_in[10];
    const float* ln1s   = (const float*)d_in[11];
    const float* ln1b   = (const float*)d_in[12];
    const float* W2     = (const float*)d_in[13];
    const float* b2     = (const float*)d_in[14];
    const float* ln2s   = (const float*)d_in[15];
    const float* ln2b   = (const float*)d_in[16];
    const float* Worb_r = (const float*)d_in[17];
    const float* Worb_i = (const float*)d_in[18];

    char* w = (char*)d_ws;
    const size_t MB = (size_t)MROWS * DDIM * sizeof(float);  // 66,846,720 B
    float* hbuf = (float*)(w);
    float* qbuf = (float*)(w + 1 * MB);
    float* kbuf = (float*)(w + 2 * MB);
    float* vbuf = (float*)(w + 3 * MB);
    float* obuf = (float*)(w + 4 * MB);
    unsigned short* packBase = (unsigned short*)(w + 5 * MB);
    const size_t packBytes = (size_t)12 * 65536 * sizeof(unsigned short);
    float2* porb     = (float2*)(w + 5 * MB + packBytes);
    float2* matA     = (float2*)((char*)porb + (size_t)NPAIR * sizeof(float2));
    float2* partials = (float2*)((char*)matA + (size_t)NEL * NEL * sizeof(float2));
    float*  lf       = (float*)((char*)partials + 256 * sizeof(float2));

    const float* bigW[6] = { Wq, Wk, Wv, Wo, W1, W2 };
    for (int l = 0; l < 2; ++l)
        for (int wi = 0; wi < 6; ++wi)
            Parton_pack_kernel<<<128, 256, 0, stream>>>(bigW[wi] + (size_t)l * 65536,
                                                        packBase + (size_t)(l * 6 + wi) * 65536);

    Parton_feat_kernel<<<MROWS, 256, 0, stream>>>(elec, W_in, hbuf);

    for (int l = 0; l < 2; ++l) {
        const unsigned short* pQ  = packBase + (size_t)(l * 6 + 0) * 65536;
        const unsigned short* pK  = packBase + (size_t)(l * 6 + 1) * 65536;
        const unsigned short* pV  = packBase + (size_t)(l * 6 + 2) * 65536;
        const unsigned short* pO  = packBase + (size_t)(l * 6 + 3) * 65536;
        const unsigned short* pW1 = packBase + (size_t)(l * 6 + 4) * 65536;
        const unsigned short* pW2 = packBase + (size_t)(l * 6 + 5) * 65536;

        Parton_gemm_kernel<<<MTILES, 256, 0, stream>>>(hbuf, pQ, bq + l * 256, nullptr, nullptr, nullptr, qbuf, 0);
        Parton_gemm_kernel<<<MTILES, 256, 0, stream>>>(hbuf, pK, bk + l * 256, nullptr, nullptr, nullptr, kbuf, 0);
        Parton_gemm_kernel<<<MTILES, 256, 0, stream>>>(hbuf, pV, bv + l * 256, nullptr, nullptr, nullptr, vbuf, 0);
        Parton_attn_kernel<<<(NPAIR * NHEAD + 255) / 256, 256, 0, stream>>>(qbuf, kbuf, vbuf, obuf);
        Parton_gemm_kernel<<<MTILES, 256, 0, stream>>>(obuf, pO, bo + l * 256, nullptr, nullptr, nullptr, qbuf, 0);
        Parton_gemm_kernel<<<MTILES, 256, 0, stream>>>(qbuf, pW1, nullptr, hbuf, ln1s + l * 256, ln1b + l * 256, hbuf, 1);
        Parton_gemm_kernel<<<MTILES, 256, 0, stream>>>(hbuf, pW2, b2 + l * 256, hbuf, ln2s + l * 256, ln2b + l * 256, hbuf, 2);
    }

    Parton_orb_kernel<<<(NPAIR + 255) / 256, 256, 0, stream>>>(hbuf, Worb_r, Worb_i, elec, porb);
    Parton_assemble_kernel<<<(NEL * NEL) / 256, 256, 0, stream>>>(elec, porb, matA, partials);
    Parton_lu_kernel<<<1, 256, 0, stream>>>(matA, lf);
    Parton_final_kernel<<<1, 256, 0, stream>>>(partials, lf, (float*)d_out, out_size);
}
